// ResNetAR_62397284876583
// MI455X (gfx1250) — matrix operand order repaired and checked by reading, not compiled, not run on hardware
//
#include <hip/hip_runtime.h>

#define EPS 1e-5f

typedef __attribute__((ext_vector_type(16))) __bf16 v16bf;
typedef __attribute__((ext_vector_type(8)))  __bf16 v8bf;
typedef __attribute__((ext_vector_type(8)))  float  v8f;
typedef __attribute__((ext_vector_type(4)))  unsigned int v4u;
typedef __attribute__((ext_vector_type(8)))  int    v8i;
typedef __attribute__((ext_vector_type(4)))  int    v4i;

#if defined(__gfx1250__) && __has_builtin(__builtin_amdgcn_tensor_load_to_lds)
#define USE_TDM 1
#else
#define USE_TDM 0
#endif

// ---------------------------------------------------------------------------
// Generic implicit-GEMM convolution using CDNA5 WMMA (bf16 in, f32 out).
//   A = weights  [Cout x K]  (K = Cin*KH*KW, natural OIHW layout)
//   B = im2col   [K x (N*OH*OW)]
// Block: 256 threads = 8 waves as 4x2 -> 64(M) x 64(N) tile, K-step 32.
// Each wave: one A fragment, two B fragments -> 2 WMMAs per K-step.
// A tile staged by the Tensor Data Mover (TDM) when available; B tile is an
// im2col gather staged transposed (BsT[col][k]) so fragments are contiguous.
// ---------------------------------------------------------------------------
#define TBM 64
#define TBN 64
#define TBK 32
#define APITCH 40   // 80B row pitch: keeps 16B fragment loads aligned

__global__ __launch_bounds__(256)
void k_conv_wmma(const __bf16* __restrict__ in,   // [Nimg, Cin, IH, IW]
                 const __bf16* __restrict__ wgt,  // [Cout, Cin*KH*KW]
                 float* __restrict__ out,
                 int Nimg, int Cin, int IH, int IW,
                 int Cout, int KH, int KW, int stride, int pad,
                 int OH, int OW, long outStride)
{
    const int Ktot = Cin * KH * KW;
    const int OHW  = OH * OW;
    const int Ntot = Nimg * OHW;

    __shared__ alignas(16) __bf16 As [TBM][APITCH];   // A tile  [m][k]
    __shared__ alignas(16) __bf16 BsT[TBN][APITCH];   // B tile transposed [n][k]

    const int tid   = threadIdx.x;
    const int lane  = tid & 31;
    const int wave  = tid >> 5;
    const int waveM = wave >> 1;   // 0..3
    const int waveN = wave & 1;    // 0..1
    const int mBase = blockIdx.y * TBM;
    const int nBase = blockIdx.x * TBN;
    const int h     = lane >> 4;   // lane half
    const int lm    = lane & 15;

    v8f acc0 = {}, acc1 = {};

    const int rB = tid & 31;        // fixed K row this thread stages for B
    const int cB0 = tid >> 5;       // first of 8 columns this thread stages

    for (int k0 = 0; k0 < Ktot; k0 += TBK) {
#if USE_TDM
        // ---- stage A tile via Tensor Data Mover (wave 0 issues the DMA) ----
        if (wave == 0) {
            unsigned ldsA = (unsigned)(unsigned long long)(void*)&As[0][0];
            unsigned long long ga =
                (unsigned long long)(const void*)(wgt + (long)mBase * Ktot + k0);
            v4u g0 = { 1u,                                   // count=1
                       ldsA,                                  // lds_addr
                       (unsigned)(ga & 0xFFFFFFFFu),          // global_addr lo
                       (unsigned)((ga >> 32) & 0x01FFFFFFu) | (2u << 30) }; // hi | type=2
            unsigned w0  = (1u << 16)    // data_size = 2 bytes
                         | (1u << 20)    // pad_enable
                         | (3u << 22)    // pad_interval: 16 DWORDs (one 64B row)
                         | (3u << 25);   // pad_amount: 4 DWORDs (16B -> 80B pitch)
            unsigned td0 = (unsigned)(Ktot - k0);    // cols remaining (OOB -> 0)
            unsigned td1 = (unsigned)(Cout - mBase); // rows remaining
            v8i g1 = { (int)w0,
                       (int)((td0 & 0xFFFFu) << 16),
                       (int)(((td0 >> 16) & 0xFFFFu) | ((td1 & 0xFFFFu) << 16)),
                       (int)(((td1 >> 16) & 0xFFFFu) | (32u << 16)),   // tile_dim0=32
                       (int)64,                                        // tile_dim1=64
                       (int)(unsigned)Ktot,                            // dim0_stride lo
                       0, 0 };
            v4i g2 = {0, 0, 0, 0}, g3 = {0, 0, 0, 0};
#if __clang_major__ >= 23
            v8i g4 = {0, 0, 0, 0, 0, 0, 0, 0};
            __builtin_amdgcn_tensor_load_to_lds(g0, g1, g2, g3, g4, 0);
#else
            __builtin_amdgcn_tensor_load_to_lds(g0, g1, g2, g3, 0);
#endif
        }
#else
        // ---- manual A tile staging: 64x32, 8 elems/thread ----
        if (k0 + TBK < Ktot)
            __builtin_prefetch(wgt + (long)mBase * Ktot + k0 + TBK, 0, 1);
        #pragma unroll
        for (int i = 0; i < 8; ++i) {
            int idx = tid + i * 256;
            int r = idx >> 5, c = idx & 31;
            int gm = mBase + r, gk = k0 + c;
            __bf16 v = (__bf16)0.0f;
            if (gm < Cout && gk < Ktot) v = wgt[(long)gm * Ktot + gk];
            As[r][c] = v;
        }
#endif
        // ---- stage B tile transposed: 32K x 64N, 8 elems/thread ----
        {
            int gk = k0 + rB;
            int ci = 0, kh = 0, kw = 0, valid = (gk < Ktot);
            if (valid) {
                ci  = gk / (KH * KW);
                int rem = gk - ci * KH * KW;
                kh  = rem / KW;
                kw  = rem - kh * KW;
            }
            #pragma unroll
            for (int i = 0; i < 8; ++i) {
                int c  = cB0 + i * 8;
                int gp = nBase + c;
                __bf16 v = (__bf16)0.0f;
                if (valid && gp < Ntot) {
                    int n  = gp / OHW;
                    int pr = gp - n * OHW;
                    int oh = pr / OW, ow = pr - (pr / OW) * OW;
                    int ih = oh * stride - pad + kh;
                    int iw = ow * stride - pad + kw;
                    if (ih >= 0 && ih < IH && iw >= 0 && iw < IW)
                        v = in[(((long)n * Cin + ci) * IH + ih) * IW + iw];
                }
                BsT[c][rB] = v;
            }
        }
#if USE_TDM
        if (wave == 0) __builtin_amdgcn_s_wait_tensorcnt(0);
#endif
        __syncthreads();

        // ---- fragments: contiguous 16B LDS reads per documented layouts ----
        const int m  = waveM * 16 + lm;
        v8bf a0 = *(const v8bf*)&As[m][h * 8];
        v8bf a1 = *(const v8bf*)&As[m][16 + h * 8];
        v16bf af = __builtin_shufflevector(a0, a1, 0,1,2,3,4,5,6,7,8,9,10,11,12,13,14,15);

        const int n0 = waveN * 32 + lm;
        v8bf b00 = *(const v8bf*)&BsT[n0][h * 8];
        v8bf b01 = *(const v8bf*)&BsT[n0][16 + h * 8];
        v16bf bf0 = __builtin_shufflevector(b00, b01, 0,1,2,3,4,5,6,7,8,9,10,11,12,13,14,15);
        v8bf b10 = *(const v8bf*)&BsT[n0 + 16][h * 8];
        v8bf b11 = *(const v8bf*)&BsT[n0 + 16][16 + h * 8];
        v16bf bf1 = __builtin_shufflevector(b10, b11, 0,1,2,3,4,5,6,7,8,9,10,11,12,13,14,15);

        acc0 = __builtin_amdgcn_wmma_f32_16x16x32_bf16(false, af, false, bf0, (short)0, acc0, false, false);
        acc1 = __builtin_amdgcn_wmma_f32_16x16x32_bf16(false, af, false, bf1, (short)0, acc1, false, false);
        __syncthreads();
    }

    // ---- store D (32-bit C/D layout: lane half selects M+8) ----
    #pragma unroll
    for (int v = 0; v < 8; ++v) {
        int co = mBase + waveM * 16 + h * 8 + v;
        if (co >= Cout) continue;
        int p0 = nBase + waveN * 32 + lm;
        int p1 = p0 + 16;
        if (p0 < Ntot) {
            int n = p0 / OHW, pr = p0 - (p0 / OHW) * OHW;
            out[(long)n * outStride + (long)co * OHW + pr] = acc0[v];
        }
        if (p1 < Ntot) {
            int n = p1 / OHW, pr = p1 - (p1 / OHW) * OHW;
            out[(long)n * outStride + (long)co * OHW + pr] = acc1[v];
        }
    }
}

// ---------------------------------------------------------------------------
// Elementwise / reduction helper kernels
// ---------------------------------------------------------------------------
__global__ void k_cast_bf16(const float* __restrict__ src, __bf16* __restrict__ dst, long n)
{
    long i = (long)blockIdx.x * blockDim.x + threadIdx.x;
    long st = (long)gridDim.x * blockDim.x;
    for (; i < n; i += st) dst[i] = (__bf16)src[i];
}

__global__ __launch_bounds__(256)
void k_bnstats(const float* __restrict__ x, float* __restrict__ mv, int N, int C, int HW)
{
    int c = blockIdx.x;
    long total = (long)N * HW;
    float s = 0.f, s2 = 0.f;
    for (long i = threadIdx.x; i < total; i += blockDim.x) {
        long n = i / HW, r = i - n * HW;
        float v = x[((long)n * C + c) * HW + r];
        s += v; s2 += v * v;
    }
    __shared__ float sh[256], sh2[256];
    sh[threadIdx.x] = s; sh2[threadIdx.x] = s2;
    __syncthreads();
    for (int o = 128; o > 0; o >>= 1) {
        if (threadIdx.x < o) { sh[threadIdx.x] += sh[threadIdx.x + o]; sh2[threadIdx.x] += sh2[threadIdx.x + o]; }
        __syncthreads();
    }
    if (threadIdx.x == 0) {
        float m = sh[0] / (float)total;
        mv[c] = m;
        mv[C + c] = sh2[0] / (float)total - m * m;
    }
}

__global__ void k_bn_act(const float* __restrict__ raw, const float* __restrict__ mv,
                         const float* __restrict__ g, const float* __restrict__ b,
                         float* outf, __bf16* outbf, int N, int C, int HW, int relu)
{
    long total = (long)N * C * HW;
    long i = (long)blockIdx.x * blockDim.x + threadIdx.x;
    long st = (long)gridDim.x * blockDim.x;
    for (; i < total; i += st) {
        int c = (int)((i / HW) % C);
        float y = g[c] * (raw[i] - mv[c]) * rsqrtf(mv[C + c] + EPS) + b[c];
        if (relu) y = fmaxf(y, 0.f);
        if (outf)  outf[i]  = y;
        if (outbf) outbf[i] = (__bf16)y;
    }
}

__global__ void k_bn_add_relu(const float* __restrict__ raw2, const float* __restrict__ mv2,
                              const float* __restrict__ g2, const float* __restrict__ b2,
                              const float* __restrict__ res, const float* __restrict__ mvds,
                              const float* __restrict__ gds, const float* __restrict__ bds,
                              float* __restrict__ outf, __bf16* __restrict__ outbf,
                              int N, int C, int HW, int hasDs)
{
    long total = (long)N * C * HW;
    long i = (long)blockIdx.x * blockDim.x + threadIdx.x;
    long st = (long)gridDim.x * blockDim.x;
    for (; i < total; i += st) {
        int c = (int)((i / HW) % C);
        float y = g2[c] * (raw2[i] - mv2[c]) * rsqrtf(mv2[C + c] + EPS) + b2[c];
        float sc;
        if (hasDs) sc = gds[c] * (res[i] - mvds[c]) * rsqrtf(mvds[C + c] + EPS) + bds[c];
        else       sc = res[i];
        float o = fmaxf(y + sc, 0.f);
        outf[i]  = o;
        outbf[i] = (__bf16)o;
    }
}

__global__ void k_maxpool(const float* __restrict__ in, float* __restrict__ outf,
                          __bf16* __restrict__ outbf, int N, int C, int IH, int IW)
{
    int OH = IH / 2, OW = IW / 2;
    long total = (long)N * C * OH * OW;
    long i = (long)blockIdx.x * blockDim.x + threadIdx.x;
    long st = (long)gridDim.x * blockDim.x;
    for (; i < total; i += st) {
        int ow = (int)(i % OW);
        int oh = (int)((i / OW) % OH);
        long nc = i / ((long)OH * OW);
        const float* p = in + nc * IH * IW;
        float m = -3.4e38f;
        for (int kh = 0; kh < 3; ++kh) {
            int ih = oh * 2 - 1 + kh;
            if (ih < 0 || ih >= IH) continue;
            for (int kw = 0; kw < 3; ++kw) {
                int iw = ow * 2 - 1 + kw;
                if (iw < 0 || iw >= IW) continue;
                m = fmaxf(m, p[ih * IW + iw]);
            }
        }
        outf[i]  = m;
        outbf[i] = (__bf16)m;
    }
}

__global__ void k_gram(const float* __restrict__ yn, float* __restrict__ A)
{
    int b = blockIdx.x / 49, hw = blockIdx.x % 49;
    int t = threadIdx.x;
    if (t >= 64) return;
    int s = t >> 3, u = t & 7;
    const float* ps = yn + ((long)(b * 8 + s) * 128) * 49 + hw;
    const float* pu = yn + ((long)(b * 8 + u) * 128) * 49 + hw;
    float d = 0.f;
    for (int c = 0; c < 128; ++c) d += ps[c * 49] * pu[c * 49];
    A[((long)(b * 49 + hw) * 8 + s) * 8 + u] = d;
}

__global__ __launch_bounds__(256)
void k_stepwc(const float* __restrict__ A, const float* __restrict__ y512,
              const float* __restrict__ yo, __bf16* __restrict__ concat, int s)
{
    int b = blockIdx.x / 49, hw = blockIdx.x % 49;
    const float* lg = A + ((long)(b * 49 + hw) * 8) * 8;   // lg[t*8 + s]
    float wts[8];
    float amax = -1e30f;
    for (int t = 0; t < s; ++t) amax = fmaxf(amax, lg[t * 8 + s]);
    float den = 0.f;
    for (int t = 0; t < 8; ++t) {
        float e = (t < s) ? __expf(lg[t * 8 + s] - amax) : 0.f;
        wts[t] = e; den += e;
    }
    float inv = 1.f / fmaxf(den, 1e-30f);
    for (int c = threadIdx.x; c < 512; c += blockDim.x) {
        float ys = y512[((long)(b * 8 + s) * 512 + c) * 49 + hw];
        float cs = 0.f;
        for (int t = 0; t < s; ++t)
            cs += wts[t] * yo[((long)(b * 8 + t) * 512 + c) * 49 + hw];
        cs *= inv;
        concat[((long)b * 1024 + c) * 49 + hw]       = (__bf16)ys;
        concat[((long)b * 1024 + 512 + c) * 49 + hw] = (__bf16)cs;
    }
}

__global__ void k_feat(const float* __restrict__ yo, const float* __restrict__ mv,
                       const float* __restrict__ g, const float* __restrict__ b,
                       float* __restrict__ feat)
{
    int i = blockIdx.x * blockDim.x + threadIdx.x;
    if (i >= 16 * 512) return;
    int n = i >> 9, c = i & 511;
    float sc = g[c] * rsqrtf(mv[512 + c] + EPS);
    float m = mv[c], bb = b[c];
    const float* p = yo + ((long)n * 512 + c) * 49;
    float sum = 0.f;
    for (int k = 0; k < 49; ++k) sum += fmaxf(sc * (p[k] - m) + bb, 0.f);
    feat[i] = sum * (1.f / 49.f);
}

__global__ __launch_bounds__(256)
void k_heads(const float* __restrict__ feat, const float* __restrict__ w1, const float* __restrict__ b1,
             const float* __restrict__ wo, const float* __restrict__ bo,
             const float* __restrict__ wa, const float* __restrict__ ba,
             float* __restrict__ off_out, float* __restrict__ ang_out)
{
    int n = blockIdx.x;                 // 0..15 == b*8+s
    __shared__ float hf[512];
    for (int j = threadIdx.x; j < 512; j += blockDim.x) {
        float a = b1[j];
        const float* f = feat + n * 512;
        const float* w = w1 + (long)j * 512;
        for (int k = 0; k < 512; ++k) a += f[k] * w[k];
        hf[j] = fmaxf(a, 0.f);
    }
    __syncthreads();
    float po = 0.f, pa = 0.f;
    for (int k = threadIdx.x; k < 512; k += blockDim.x) { po += hf[k] * wo[k]; pa += hf[k] * wa[k]; }
    __shared__ float ro[256], ra[256];
    ro[threadIdx.x] = po; ra[threadIdx.x] = pa;
    __syncthreads();
    for (int o = 128; o > 0; o >>= 1) {
        if (threadIdx.x < o) { ro[threadIdx.x] += ro[threadIdx.x + o]; ra[threadIdx.x] += ra[threadIdx.x + o]; }
        __syncthreads();
    }
    if (threadIdx.x == 0) { off_out[n] = ro[0] + bo[0]; ang_out[n] = ra[0] + ba[0]; }
}

// ---------------------------------------------------------------------------
// Host orchestration
// ---------------------------------------------------------------------------
static void launch_conv(const __bf16* in, const __bf16* w, float* out,
                        int Nimg, int Cin, int IH, int IW,
                        int Cout, int KH, int KW, int stride, int pad,
                        long outStride, hipStream_t s)
{
    int OH = (IH + 2 * pad - KH) / stride + 1;
    int OW = (IW + 2 * pad - KW) / stride + 1;
    long Ntot = (long)Nimg * OH * OW;
    dim3 g((unsigned)((Ntot + TBN - 1) / TBN), (unsigned)((Cout + TBM - 1) / TBM));
    k_conv_wmma<<<g, 256, 0, s>>>(in, w, out, Nimg, Cin, IH, IW, Cout, KH, KW,
                                  stride, pad, OH, OW, outStride);
}

static void cast_w(const float* src, __bf16* dst, long n, hipStream_t s)
{
    int blocks = (int)((n + 255) / 256);
    if (blocks > 4096) blocks = 4096;
    k_cast_bf16<<<blocks, 256, 0, s>>>(src, dst, n);
}

extern "C" void kernel_launch(void* const* d_in, const int* in_sizes, int n_in,
                              void* d_out, int out_size, void* d_ws, size_t ws_size,
                              hipStream_t stream)
{
    (void)in_sizes; (void)n_in; (void)out_size; (void)ws_size;
    const int NI = 16;                       // B*S images

    int ii = 0;
    auto nxt = [&]() -> const float* { return (const float*)d_in[ii++]; };
    const float* x       = nxt();
    const float* conv1w  = nxt();
    const float* bn1_g   = nxt();
    const float* bn1_b   = nxt();

    struct Blk { const float *w1,*g1,*b1,*w2,*g2,*b2,*dsw,*dsg,*dsb; int hasDs; };
    Blk blks[4][2];
    const int  chs[5]    = {64, 64, 128, 256, 512};
    const int  lstride[4]= {1, 2, 2, 2};
    for (int L = 0; L < 4; ++L)
        for (int k = 0; k < 2; ++k) {
            Blk& B = blks[L][k];
            B.w1 = nxt(); B.g1 = nxt(); B.b1 = nxt();
            B.w2 = nxt(); B.g2 = nxt(); B.b2 = nxt();
            B.hasDs = (k == 0) && (lstride[L] != 1 || chs[L] != chs[L + 1]);
            if (B.hasDs) { B.dsw = nxt(); B.dsg = nxt(); B.dsb = nxt(); }
            else         { B.dsw = B.dsg = B.dsb = nullptr; }
        }
    const float* projw   = nxt();
    const float* endw    = nxt();
    const float* bnend_g = nxt();
    const float* bnend_b = nxt();
    const float* fc_w    = nxt();
    const float* fc_b    = nxt();
    const float* fco_w   = nxt();
    const float* fco_b   = nxt();
    const float* fca_w   = nxt();
    const float* fca_b   = nxt();

    char* base = (char*)d_ws;
    size_t off = 0;
    auto alloc = [&](size_t bytes) -> char* {
        size_t cur = (off + 255) & ~(size_t)255;
        off = cur + bytes;
        return base + cur;
    };

    const long E1 = (long)NI * 64 * 56 * 56;
    const long R1 = (long)NI * 64 * 112 * 112;

    __bf16* xbf   = (__bf16*)alloc(2408448L * 2);
    float*  c1raw = (float*)alloc(R1 * 4);
    float*  curF  = (float*)alloc(E1 * 4);
    __bf16* curB  = (__bf16*)alloc(E1 * 2);
    float*  t1raw = (float*)alloc(E1 * 4);
    __bf16* t1bf  = (__bf16*)alloc(E1 * 2);
    float*  t2raw = (float*)alloc(E1 * 4);
    float*  dsraw = (float*)alloc(E1 * 4);
    float*  outF  = (float*)alloc(E1 * 4);
    __bf16* outB  = (__bf16*)alloc(E1 * 2);
    float*  mvA   = (float*)alloc(1024 * 4);
    float*  mvB   = (float*)alloc(1024 * 4);
    float*  Agram = (float*)alloc(2L * 49 * 64 * 4);
    __bf16* concat= (__bf16*)alloc(2L * 1024 * 49 * 2);
    float*  feat  = (float*)alloc(16L * 512 * 4);

    auto walloc = [&](long n) -> __bf16* { return (__bf16*)alloc(n * 2); };
    __bf16* conv1wb = walloc(64L * 3 * 49);   cast_w(conv1w, conv1wb, 64L * 3 * 49, stream);
    __bf16* w1b[4][2]; __bf16* w2b[4][2]; __bf16* dswb[4];
    for (int L = 0; L < 4; ++L) {
        int ci = chs[L], co = chs[L + 1];
        for (int k = 0; k < 2; ++k) {
            int cin = (k == 0) ? ci : co;
            long n1 = (long)co * cin * 9, n2 = (long)co * co * 9;
            w1b[L][k] = walloc(n1); cast_w(blks[L][k].w1, w1b[L][k], n1, stream);
            w2b[L][k] = walloc(n2); cast_w(blks[L][k].w2, w2b[L][k], n2, stream);
        }
        if (blks[L][0].hasDs) {
            long nd = (long)co * ci;
            dswb[L] = walloc(nd); cast_w(blks[L][0].dsw, dswb[L], nd, stream);
        } else dswb[L] = nullptr;
    }
    __bf16* projwb = walloc(128L * 512);  cast_w(projw, projwb, 128L * 512, stream);
    __bf16* endwb  = walloc(512L * 1024); cast_w(endw,  endwb,  512L * 1024, stream);

    // ---- stem ----
    cast_w(x, xbf, 2408448L, stream);
    launch_conv(xbf, conv1wb, c1raw, NI, 3, 224, 224, 64, 7, 7, 2, 3, 64L * 112 * 112, stream);
    k_bnstats<<<64, 256, 0, stream>>>(c1raw, mvA, NI, 64, 112 * 112);
    k_bn_act<<<(unsigned)((R1 + 255) / 256), 256, 0, stream>>>(
        c1raw, mvA, bn1_g, bn1_b, c1raw, (__bf16*)nullptr, NI, 64, 112 * 112, 1);
    k_maxpool<<<(unsigned)((E1 + 255) / 256), 256, 0, stream>>>(c1raw, curF, curB, NI, 64, 112, 112);

    // ---- residual layers ----
    int ih = 56;
    for (int L = 0; L < 4; ++L) {
        int ci = chs[L], co = chs[L + 1], st = lstride[L];
        int oh = ih / st;
        for (int k = 0; k < 2; ++k) {
            const Blk& B = blks[L][k];
            int cin = (k == 0) ? ci : co;
            int sIn = (k == 0) ? st : 1;
            int inH = (k == 0) ? ih : oh;
            long hw  = (long)oh * oh;
            long tot = (long)NI * co * hw;
            unsigned eb = (unsigned)((tot + 255) / 256);

            launch_conv(curB, w1b[L][k], t1raw, NI, cin, inH, inH, co, 3, 3, sIn, 1, (long)co * hw, stream);
            k_bnstats<<<co, 256, 0, stream>>>(t1raw, mvA, NI, co, (int)hw);
            k_bn_act<<<eb, 256, 0, stream>>>(t1raw, mvA, B.g1, B.b1, t1raw, t1bf, NI, co, (int)hw, 1);

            launch_conv(t1bf, w2b[L][k], t2raw, NI, co, oh, oh, co, 3, 3, 1, 1, (long)co * hw, stream);
            k_bnstats<<<co, 256, 0, stream>>>(t2raw, mvA, NI, co, (int)hw);

            if (B.hasDs) {
                launch_conv(curB, dswb[L], dsraw, NI, cin, inH, inH, co, 1, 1, sIn, 0, (long)co * hw, stream);
                k_bnstats<<<co, 256, 0, stream>>>(dsraw, mvB, NI, co, (int)hw);
                k_bn_add_relu<<<eb, 256, 0, stream>>>(t2raw, mvA, B.g2, B.b2,
                                                      dsraw, mvB, B.dsg, B.dsb,
                                                      outF, outB, NI, co, (int)hw, 1);
            } else {
                k_bn_add_relu<<<eb, 256, 0, stream>>>(t2raw, mvA, B.g2, B.b2,
                                                      curF, nullptr, nullptr, nullptr,
                                                      outF, outB, NI, co, (int)hw, 0);
            }
            { float*  tf = curF; curF = outF; outF = tf; }
            { __bf16* tb = curB; curB = outB; outB = tb; }
        }
        ih = oh;
    }

    // ---- output regions ----
    float* out_off = (float*)d_out;
    float* out_ang = out_off + 16;
    float* yo      = out_ang + 16;           // [2,8,512,49]
    float* ynorm   = yo + 2L * 8 * 512 * 49; // [2,8,128,49]

    launch_conv(curB, projwb, ynorm, NI, 512, 7, 7, 128, 1, 1, 1, 0, 128L * 49, stream);
    k_gram<<<98, 64, 0, stream>>>(ynorm, Agram);

    for (int s = 0; s < 8; ++s) {
        k_stepwc<<<98, 256, 0, stream>>>(Agram, curF, yo, concat, s);
        launch_conv(concat, endwb, yo + (long)s * 512 * 49,
                    2, 1024, 7, 7, 512, 1, 1, 1, 0, 8L * 512 * 49, stream);
    }

    k_bnstats<<<512, 256, 0, stream>>>(yo, mvA, 16, 512, 49);
    k_feat<<<(16 * 512 + 255) / 256, 256, 0, stream>>>(yo, mvA, bnend_g, bnend_b, feat);
    k_heads<<<16, 256, 0, stream>>>(feat, fc_w, fc_b, fco_w, fco_b, fca_w, fca_b, out_off, out_ang);
}
